// AutoRegressiveDistribution_78134045049550
// MI455X (gfx1250) — matrix operand order repaired and checked by reading, not compiled, not run on hardware
//
#include <hip/hip_runtime.h>
#include <hip/hip_bf16.h>

#define DD 64
#define HH 512
#define CC 256
#define BCTX 2048
#define NROWS 32768      // S*B = 16*2048
#define ROWS 32          // rows per workgroup
#define H_PAD 520        // ushort stride for h1/h2 (1040B -> +4 banks per row)

typedef __attribute__((ext_vector_type(16))) __bf16 v16bf;
typedef __attribute__((ext_vector_type(8)))  float  v8f;

union Frag16 { v16bf v; uint4 u[2]; };

__device__ __forceinline__ unsigned short f2bf(float f) {
  unsigned u = __float_as_uint(f);
  u += 0x7FFFu + ((u >> 16) & 1u);          // round-to-nearest-even
  return (unsigned short)(u >> 16);
}
__device__ __forceinline__ float bf2f(unsigned h) {
  return __uint_as_float(h << 16);
}

// ---------------- prep kernel 1: ctx_proj = context @ Wc + b1 ----------------
__global__ __launch_bounds__(256) void ctx_proj_kernel(
    const float* __restrict__ context, const float* __restrict__ Wc,
    const float* __restrict__ b1, float* __restrict__ ctxp) {
  int idx = blockIdx.x * 256 + threadIdx.x;   // over 2048*512
  int b = idx >> 9;
  int h = idx & 511;
  float acc = b1[h];
  const float* crow = context + b * CC;
  #pragma unroll 4
  for (int c = 0; c < CC; ++c) acc += crow[c] * Wc[c * HH + h];
  ctxp[idx] = acc;
}

// ---------------- prep kernel 2: masked weights -----------------------------
__global__ __launch_bounds__(256) void prep_weights_kernel(
    const float* __restrict__ W1, const float* __restrict__ W2,
    const float* __restrict__ W3, float* __restrict__ MW1,
    unsigned short* __restrict__ MW2T, float* __restrict__ MW3T) {
  int idx = blockIdx.x * 256 + threadIdx.x;   // over 512*512
  // MW1 (64x512): M1[i][j] = hid_deg[j] >= in_deg[i]
  if (idx < DD * HH) {
    int i = idx / HH, j = idx % HH;
    int hd = j % (DD - 1) + 1;
    MW1[idx] = (hd >= i + 1) ? W1[idx] : 0.0f;
  }
  // MW2T (col-major of masked W2): MW2T[n][k] = (hid_deg[n]>=hid_deg[k]) * W2[k][n]
  {
    int n = idx / HH, k = idx % HH;
    int hdn = n % (DD - 1) + 1, hdk = k % (DD - 1) + 1;
    float v = (hdn >= hdk) ? W2[k * HH + n] : 0.0f;
    MW2T[idx] = f2bf(v);
  }
  // MW3T (128x512): MW3T[o][j] = (out_deg[o] > hid_deg[j]) * W3[j][o]
  if (idx < 2 * DD * HH) {
    int o = idx / HH, j = idx % HH;
    int hdj = j % (DD - 1) + 1;
    float v = ((o % DD) + 1 > hdj) ? W3[j * (2 * DD) + o] : 0.0f;
    MW3T[idx] = v;
  }
}

// ---------------- main persistent autoregressive kernel ---------------------
__global__ __launch_bounds__(256) void made_sample_kernel(
    const float* __restrict__ ctxp, const float* __restrict__ eps,
    const float* __restrict__ MW1, const unsigned short* __restrict__ MW2T,
    const float* __restrict__ MW3T, const float* __restrict__ b2,
    const float* __restrict__ b3, float* __restrict__ out) {
  __shared__ unsigned short h1[ROWS * H_PAD];   // bf16 activations (GEMM A)
  __shared__ unsigned short h2[ROWS * H_PAD];   // bf16 hidden2
  __shared__ float b2s[HH];
  __shared__ float b3s[2 * DD];
  __shared__ float mw1d[HH];
  __shared__ float mw3a[HH];
  __shared__ float mw3b[HH];
  __shared__ float zds[ROWS];

  const int tid  = threadIdx.x;
  const int lane = tid & 31;
  const int wave = tid >> 5;
  const int R0   = blockIdx.x * ROWS;

  for (int i = tid; i < HH; i += 256) b2s[i] = b2[i];
  if (tid < 2 * DD) b3s[tid] = b3[tid];

  // register-resident a1 chunk: this thread owns row (tid>>3), cols j0..j0+63
  const int myrow = tid >> 3;
  const int j0    = (tid & 7) * 64;
  const int grow  = R0 + myrow;                 // global row = s*B + b
  float a1[64];
  {
    const float4* src =
        (const float4*)(ctxp + (size_t)(grow & (BCTX - 1)) * HH + j0);
    #pragma unroll
    for (int i = 0; i < 16; ++i) {
      float4 v = src[i];
      a1[4 * i + 0] = v.x; a1[4 * i + 1] = v.y;
      a1[4 * i + 2] = v.z; a1[4 * i + 3] = v.w;
    }
  }

  const int kchunk = lane >> 4;   // 0 or 1: which K-half this lane holds
  const int lrow   = lane & 15;

  for (int d = 0; d < DD; ++d) {
    // per-step vector preloads (W1 row d, MW3T cols d and D+d)
    for (int i = tid; i < HH; i += 256) {
      mw1d[i] = MW1[d * HH + i];
      mw3a[i] = MW3T[d * HH + i];
      mw3b[i] = MW3T[(DD + d) * HH + i];
    }
    // ---- phase A: h1 = bf16(relu(a1)) ----
    {
      unsigned* dst = (unsigned*)(h1 + myrow * H_PAD + j0);
      #pragma unroll
      for (int i = 0; i < 32; ++i) {
        float x0 = fmaxf(a1[2 * i],     0.0f);
        float x1 = fmaxf(a1[2 * i + 1], 0.0f);
        dst[i] = (unsigned)f2bf(x0) | ((unsigned)f2bf(x1) << 16);
      }
    }
    __syncthreads();

    // ---- phase B: h2 = relu(h1 @ MW2 + b2) via WMMA bf16 ----
    for (int t = 0; t < 4; ++t) {
      const int n0 = (wave * 4 + t) * 16;
      v8f acc0 = {}; v8f acc1 = {};
      const unsigned short* bbase  = MW2T + (size_t)(n0 + lrow) * HH + kchunk * 8;
      const unsigned short* a0base = h1 + lrow * H_PAD + kchunk * 8;
      const unsigned short* a1base = h1 + (16 + lrow) * H_PAD + kchunk * 8;
      #pragma unroll 4
      for (int k0 = 0; k0 < HH; k0 += 32) {
        Frag16 fa0, fa1, fb;
        fa0.u[0] = *(const uint4*)(a0base + k0);        // K = k0+kc*8 .. +7
        fa0.u[1] = *(const uint4*)(a0base + k0 + 16);   // K = k0+16+kc*8 .. +7
        fa1.u[0] = *(const uint4*)(a1base + k0);
        fa1.u[1] = *(const uint4*)(a1base + k0 + 16);
        fb.u[0]  = *(const uint4*)(bbase + k0);         // K = k0+kc*8 .. +7
        fb.u[1]  = *(const uint4*)(bbase + k0 + 16);    // K = k0+16+kc*8 .. +7
        acc0 = __builtin_amdgcn_wmma_f32_16x16x32_bf16(
            false, fa0.v, false, fb.v, (short)0, acc0, false, false);
        acc1 = __builtin_amdgcn_wmma_f32_16x16x32_bf16(
            false, fa1.v, false, fb.v, (short)0, acc1, false, false);
      }
      // epilogue: +b2, relu, ->bf16 LDS. C layout: VGPR v -> M = v + (lane/16)*8
      const int erow = kchunk * 8;
      const int ecol = n0 + lrow;
      const float bias = b2s[ecol];
      #pragma unroll
      for (int v = 0; v < 8; ++v) {
        float x0 = fmaxf(acc0[v] + bias, 0.0f);
        float x1 = fmaxf(acc1[v] + bias, 0.0f);
        h2[(erow + v) * H_PAD + ecol]      = f2bf(x0);
        h2[(16 + erow + v) * H_PAD + ecol] = f2bf(x1);
      }
    }
    __syncthreads();

    // ---- phase C: mu_d / pre_scale_d dot products (8 threads per row) ----
    float mu = 0.0f, ps = 0.0f;
    {
      const unsigned* hrow = (const unsigned*)(h2 + myrow * H_PAD + j0);
      #pragma unroll 8
      for (int i = 0; i < 32; ++i) {
        unsigned pk = hrow[i];
        float v0 = bf2f(pk & 0xFFFFu);
        float v1 = bf2f(pk >> 16);
        mu += v0 * mw3a[j0 + 2 * i] + v1 * mw3a[j0 + 2 * i + 1];
        ps += v0 * mw3b[j0 + 2 * i] + v1 * mw3b[j0 + 2 * i + 1];
      }
    }
    #pragma unroll
    for (int off = 4; off >= 1; off >>= 1) {
      mu += __shfl_xor(mu, off, 32);
      ps += __shfl_xor(ps, off, 32);
    }
    if ((tid & 7) == 0) {
      mu += b3s[d];
      ps += b3s[DD + d];
      float sp = fmaxf(ps, 0.0f) + log1pf(__expf(-fabsf(ps)));  // softplus
      float ed = eps[(size_t)grow * DD + d];
      float z  = mu + sp * ed;
      out[(size_t)grow * DD + d]                          = z;   // z_samples
      out[(size_t)NROWS * DD + (size_t)grow * DD + d]     = mu;  // mus
      out[2 * (size_t)NROWS * DD + (size_t)grow * DD + d] = sp;  // scales
      zds[myrow] = z;
    }
    __syncthreads();

    // ---- phase D: rank-1 update a1 += z_d * MW1[d,:] ----
    {
      float zr = zds[myrow];
      #pragma unroll
      for (int i = 0; i < 64; ++i) a1[i] += zr * mw1d[j0 + i];
    }
    __syncthreads();
  }
}

extern "C" void kernel_launch(void* const* d_in, const int* in_sizes, int n_in,
                              void* d_out, int out_size, void* d_ws, size_t ws_size,
                              hipStream_t stream) {
  (void)in_sizes; (void)n_in; (void)out_size; (void)ws_size;
  const float* context = (const float*)d_in[0];
  const float* eps     = (const float*)d_in[1];
  const float* W1      = (const float*)d_in[2];
  const float* Wc      = (const float*)d_in[3];
  const float* b1      = (const float*)d_in[4];
  const float* W2      = (const float*)d_in[5];
  const float* b2      = (const float*)d_in[6];
  const float* W3      = (const float*)d_in[7];
  const float* b3      = (const float*)d_in[8];
  float* out = (float*)d_out;

  char* ws = (char*)d_ws;
  size_t off = 0;
  float* ctxp = (float*)(ws + off);           off += (size_t)BCTX * HH * 4;  // 4 MB
  float* MW1  = (float*)(ws + off);           off += (size_t)DD * HH * 4;    // 128 KB
  float* MW3T = (float*)(ws + off);           off += (size_t)2 * DD * HH * 4;// 256 KB
  unsigned short* MW2T = (unsigned short*)(ws + off);                        // 512 KB

  ctx_proj_kernel<<<(BCTX * HH) / 256, 256, 0, stream>>>(context, Wc, b1, ctxp);
  prep_weights_kernel<<<(HH * HH) / 256, 256, 0, stream>>>(W1, W2, W3, MW1, MW2T, MW3T);
  made_sample_kernel<<<NROWS / ROWS, 256, 0, stream>>>(ctxp, eps, MW1, MW2T, MW3T,
                                                       b2, b3, out);
}
